// SegmNet_59588376264890
// MI455X (gfx1250) — matrix operand order repaired and checked by reading, not compiled, not run on hardware
//
#include <hip/hip_runtime.h>
#include <hip/hip_bf16.h>

// ---------------------------------------------------------------------------
// CDNA5 (gfx1250) implementation. wave32, WMMA f32_16x16x32_f16 for all
// GEMM-shaped work (1x1/3x3 convs via implicit GEMM, similarity GEMM).
// All channel/spatial dims are powers of two -> shifts/masks, branchless
// clamped gathers (unconditional loads + cndmask), 32-bit voffsets.
// ---------------------------------------------------------------------------

typedef __attribute__((ext_vector_type(16))) _Float16 v16h;
typedef __attribute__((ext_vector_type(8)))  float    v8f;

union HalfPair { unsigned int u; _Float16 h[2]; };

// Branchless top-3 insert: 3-deep sorting network (5 min/max VALU ops).
__device__ __forceinline__ void ins3(float v, float& t0, float& t1, float& t2) {
  float m0 = fmaxf(t0, v);
  float s0 = fminf(t0, v);
  float m1 = fmaxf(t1, s0);
  float s1 = fminf(t1, s0);
  float m2 = fmaxf(t2, s1);
  t0 = m0; t1 = m1; t2 = m2;
}

// ---------------------------------------------------------------------------
// Implicit-GEMM conv (NCHW, OIHW) + BN (+ optional ReLU), f16 WMMA, f32 acc.
// Block = 128 threads = 4 waves. Each wave: one 16(co) x 16(spatial) tile.
// grid = (HW/64, ceil(co/16), B).  H == W == 1<<logW; ci, co powers of 2.
// KSZ template const -> const-divisor k decomposition. A staged in LDS in the
// [m][k] order; lanes take k = 16*hi + s (contiguous order). B gather: clamped address + select (no branches).
// ---------------------------------------------------------------------------
template <int KSZ>
__global__ void __launch_bounds__(128)
convbn_wmma(const float* __restrict__ in, const float* __restrict__ wgt,
            const float* __restrict__ bias, const float* __restrict__ g,
            const float* __restrict__ be, const float* __restrict__ rm,
            const float* __restrict__ rv, float* __restrict__ out,
            int ci, int co, int logW, int pad, int relu)
{
  constexpr int KK2 = KSZ * KSZ;
  __shared__ _Float16 ldsA[16][32];   // [m][k] for current 32-wide K slab

  const int tid    = threadIdx.x;
  const int lane   = tid & 31;
  const int waveId = tid >> 5;
  const int laneLo = lane & 15;
  const int hi     = lane >> 4;       // 0/1: which K half-group this lane owns
  const int b      = blockIdx.z;
  const int mbase  = blockIdx.y * 16;
  const int W      = 1 << logW;
  const int wMask  = W - 1;
  const int ciMask = ci - 1;
  const int coMask = co - 1;
  const int n      = blockIdx.x * 64 + waveId * 16 + laneLo;  // spatial col
  const int oy     = n >> logW;
  const int ox     = n & wMask;
  const int K      = ci * KK2;
  const float* inB = in + ((size_t)b * ci << (2 * logW));

  v8f acc = {};
  for (int kk = 0; kk < K; kk += 32) {
    __syncthreads();
    // ---- cooperative stage of A tile (shared by all 4 waves) ----
    {
      int m  = tid >> 3;            // 0..15
      int k0 = (tid & 7) * 4;       // 0..28
      int mm = mbase + m;
      bool mok = (mbase + m) < co;
      int mmc = mm & coMask;
      #pragma unroll
      for (int j = 0; j < 4; ++j) {
        int kidx = kk + k0 + j;
        bool ok = mok & (kidx < K);
        int kc = min(kidx, K - 1);
        unsigned c = (unsigned)kc / KK2;            // const divisor
        unsigned r = (unsigned)kc - c * KK2;
        float v = wgt[(unsigned)((mmc * ci + c) * KK2 + r)];
        ldsA[m][k0 + j] = (_Float16)(ok ? v : 0.0f);
      }
    }
    __syncthreads();

    // ---- A operand from LDS (32-bit paired reads, layout-exact) ----
    v16h aOp;
    #pragma unroll
    for (int v = 0; v < 8; ++v) {
      int k0 = hi * 16 + 2 * v;
      HalfPair p;
      p.u = *(const unsigned int*)&ldsA[laneLo][k0];
      aOp[2 * v]     = p.h[0];
      aOp[2 * v + 1] = p.h[1];
    }

    // ---- B operand: branchless im2col gather (lane = N column) ----
    v16h bOp;
    #pragma unroll
    for (int e = 0; e < 16; ++e) {
      int kidx = kk + hi * 16 + e;
      float v;
      if (KSZ == 1) {
        unsigned off = ((unsigned)(kidx & ciMask) << (2 * logW)) + n;
        float raw = inB[off];
        v = (kidx < K) ? raw : 0.0f;
      } else {
        int kc = min(kidx, K - 1);
        unsigned c = (unsigned)kc / KK2;            // const divisor (9)
        unsigned r = (unsigned)kc - c * KK2;
        int kh = (int)(r / KSZ);                    // const divisor (3)
        int kw = (int)(r - kh * KSZ);
        int iy = oy + kh - pad, ix = ox + kw - pad;
        bool ok = (kidx < K) & ((unsigned)iy < (unsigned)W) &
                  ((unsigned)ix < (unsigned)W);
        unsigned off = ((c & (unsigned)ciMask) << (2 * logW)) +
                       ((unsigned)(iy & wMask) << logW) + (unsigned)(ix & wMask);
        float raw = inB[off];                       // always in-bounds
        v = ok ? raw : 0.0f;
      }
      bOp[e] = (_Float16)v;
    }

    // prefetch next K-slab's input row into caches
    if (kk + 32 < K) {
      unsigned cn = (unsigned)(kk + 32) / KK2;
      __builtin_prefetch(&inB[(cn << (2 * logW)) + (unsigned)n], 0, 1);
    }

    acc = __builtin_amdgcn_wmma_f32_16x16x32_f16(false, aOp, false, bOp,
                                                 (short)0, acc, false, false);
  }

  // ---- fused bias + BN (+ReLU) and store. C/D layout: m = v + 8*hi ----
  float* outB = out + ((size_t)b * co << (2 * logW));
  #pragma unroll
  for (int v = 0; v < 8; ++v) {
    int m = mbase + v + hi * 8;
    if (m < co) {
      float sc = g[m] * rsqrtf(rv[m] + 1e-5f);
      float y  = (acc[v] + bias[m] - rm[m]) * sc + be[m];
      if (relu) y = fmaxf(y, 0.0f);
      outB[((unsigned)m << (2 * logW)) + (unsigned)n] = y;
    }
  }
}

// ---------------------------------------------------------------------------
// L2-normalize over 64 channels at each (b, q=spatial) and pack to f16.
// transposed=1: out[(b*1024+q)*32 + c2]  (A operand, row-major q x c pairs)
// transposed=0: out[(b*32+c2)*1024 + q]  (B operand, channel-pair x t)
// ---------------------------------------------------------------------------
__global__ void l2pack(const float* __restrict__ f, unsigned int* __restrict__ out,
                       int transposed)
{
  int idx = blockIdx.x * blockDim.x + threadIdx.x;
  if (idx >= 8 * 1024) return;
  int b = idx >> 10, q = idx & 1023;
  const float* p = f + ((size_t)b << 16) + q;   // b*64*1024
  float vals[64];
  float ss = 0.0f;
  #pragma unroll
  for (int c = 0; c < 64; ++c) { float v = p[(unsigned)c << 10]; vals[c] = v; ss += v * v; }
  float inv = 1.0f / fmaxf(sqrtf(ss), 1e-12f);
  unsigned int* po = out + ((size_t)b << 15);   // b*32*1024
  #pragma unroll
  for (int c2 = 0; c2 < 32; ++c2) {
    HalfPair hp;
    hp.h[0] = (_Float16)(vals[2 * c2] * inv);
    hp.h[1] = (_Float16)(vals[2 * c2 + 1] * inv);
    if (transposed) po[(unsigned)(q << 5) + c2] = hp.u;
    else            po[((unsigned)c2 << 10) + q] = hp.u;
  }
}

// ---------------------------------------------------------------------------
// sim[b,q,t] = <ftest_n[b,:,q], ftrain_n[b,:,t]> (K=64) via 2x WMMA per tile,
// streaming masked top-3 (pos/neg) per q (branchless sorting-network insert),
// cross-lane merge in LDS, fused softmax -> segm_layers (B,2,32,32).
// One wave per 16-q block, block = 4 waves. grid = (16, B)
// ---------------------------------------------------------------------------
__global__ void __launch_bounds__(128)
simtopk_kernel(const unsigned int* __restrict__ aT,   // [b][q][32] half2
               const unsigned int* __restrict__ bP,   // [b][c2=32][1024] half2
               const float* __restrict__ mask,        // (B,1,512,512)
               float* __restrict__ segm)              // (B,2,32,32)
{
  __shared__ float red[4][16][16][6];  // [wave][q][tlane][tp0..2, tn0..2]

  const int tid = threadIdx.x;
  const int lane = tid & 31, wave = tid >> 5;
  const int laneLo = lane & 15, hi = lane >> 4;
  const int b = blockIdx.y;
  const int qtile = blockIdx.x * 4 + wave;
  const int q = qtile * 16 + laneLo;

  const unsigned int* aB = aT + ((size_t)b << 15);  // b*1024*32
  const unsigned int* bB = bP + ((size_t)b << 15);  // b*32*1024
  const float* mB = mask + ((size_t)b << 18);       // b*512*512

  // A operand (invariant over the t loop): K = c = 0..63 -> two v16h
  v16h aLo, aHi;
  #pragma unroll
  for (int v = 0; v < 8; ++v) {
    int c2 = hi * 8 + v;
    HalfPair p;
    p.u = aB[(unsigned)(q << 5) + c2];
    aLo[2 * v] = p.h[0]; aLo[2 * v + 1] = p.h[1];
    p.u = aB[(unsigned)(q << 5) + c2 + 16];
    aHi[2 * v] = p.h[0]; aHi[2 * v + 1] = p.h[1];
  }

  float tp0[8], tp1[8], tp2[8], tn0[8], tn1[8], tn2[8];
  #pragma unroll
  for (int v = 0; v < 8; ++v) {
    tp0[v] = tp1[v] = tp2[v] = -1e30f;
    tn0[v] = tn1[v] = tn2[v] = -1e30f;
  }

  for (int tt = 0; tt < 64; ++tt) {
    int n = tt * 16 + laneLo;
    v16h bLo, bHi;
    #pragma unroll
    for (int v = 0; v < 8; ++v) {
      int c2 = hi * 8 + v;
      HalfPair p;
      p.u = bB[((unsigned)c2 << 10) + n];
      bLo[2 * v] = p.h[0]; bLo[2 * v + 1] = p.h[1];
      p.u = bB[((unsigned)(c2 + 16) << 10) + n];
      bHi[2 * v] = p.h[0]; bHi[2 * v + 1] = p.h[1];
    }
    v8f acc = {};
    acc = __builtin_amdgcn_wmma_f32_16x16x32_f16(false, aLo, false, bLo,
                                                 (short)0, acc, false, false);
    acc = __builtin_amdgcn_wmma_f32_16x16x32_f16(false, aHi, false, bHi,
                                                 (short)0, acc, false, false);
    // mask_train[:, :, ::16, ::16] at t = n (t grid is 32x32)
    int ty = n >> 5, tx = n & 31;
    float mp = mB[(unsigned)((ty << 13) + (tx << 4))];  // (ty*16)*512 + tx*16
    float mn = 1.0f - mp;
    #pragma unroll
    for (int v = 0; v < 8; ++v) {
      float s = acc[v];
      ins3(s * mp, tp0[v], tp1[v], tp2[v]);
      ins3(s * mn, tn0[v], tn1[v], tn2[v]);
    }
  }

  #pragma unroll
  for (int v = 0; v < 8; ++v) {
    int ql = v + hi * 8;
    red[wave][ql][laneLo][0] = tp0[v];
    red[wave][ql][laneLo][1] = tp1[v];
    red[wave][ql][laneLo][2] = tp2[v];
    red[wave][ql][laneLo][3] = tn0[v];
    red[wave][ql][laneLo][4] = tn1[v];
    red[wave][ql][laneLo][5] = tn2[v];
  }
  __syncthreads();

  if (lane < 16) {
    int ql = lane;
    float p0 = -1e30f, p1 = -1e30f, p2 = -1e30f;
    float n0 = -1e30f, n1 = -1e30f, n2 = -1e30f;
    #pragma unroll
    for (int j = 0; j < 16; ++j) {
      ins3(red[wave][ql][j][0], p0, p1, p2);
      ins3(red[wave][ql][j][1], p0, p1, p2);
      ins3(red[wave][ql][j][2], p0, p1, p2);
      ins3(red[wave][ql][j][3], n0, n1, n2);
      ins3(red[wave][ql][j][4], n0, n1, n2);
      ins3(red[wave][ql][j][5], n0, n1, n2);
    }
    float pp = (p0 + p1 + p2) * (1.0f / 3.0f);
    float pn = (n0 + n1 + n2) * (1.0f / 3.0f);
    float mx = fmaxf(pp, pn);
    float e0 = __expf(pp - mx), e1 = __expf(pn - mx);
    float sm0 = e0 / (e0 + e1);
    int qg = qtile * 16 + ql;
    segm[(((size_t)b * 2 + 0) << 10) + qg] = sm0;  // softmax pos channel
    segm[(((size_t)b * 2 + 1) << 10) + qg] = pp;   // raw pred_pos channel
  }
}

// ---------------------------------------------------------------------------
// Bilinear x2 upsample (half-pixel centers), optional pre-add skip, opt ReLU.
// out: (8, C, 2H, 2W) from a(+b): (8, C, H, W).  H == W == 1<<logW, C == 1<<logC.
// x2 bilinear closed form: y0 = (oy-1)>>1 (arith), fy = oy&1 ? 0.25 : 0.75.
// ---------------------------------------------------------------------------
__global__ void up2_kernel(const float* __restrict__ a, const float* __restrict__ bopt,
                           float* __restrict__ out, int logC, int logW, int relu)
{
  const int W = 1 << logW;
  const int logOW = logW + 1;
  size_t total = (size_t)8 << (logC + 2 * logOW);
  size_t idx = (size_t)blockIdx.x * blockDim.x + threadIdx.x;
  if (idx >= total) return;

  int ox = (int)(idx & ((2 * W) - 1));
  int oy = (int)((idx >> logOW) & ((2 * W) - 1));
  int c  = (int)((idx >> (2 * logOW)) & ((1 << logC) - 1));
  int bb = (int)(idx >> (2 * logOW + logC));

  int y0 = (oy - 1) >> 1;             // arithmetic shift: oy=0 -> -1
  int x0 = (ox - 1) >> 1;
  float fy = (oy & 1) ? 0.25f : 0.75f;
  float fx = (ox & 1) ? 0.25f : 0.75f;
  int y1 = min(y0 + 1, W - 1), x1 = min(x0 + 1, W - 1);
  y0 = max(y0, 0);  x0 = max(x0, 0);

  size_t plane = ((((size_t)bb << logC) + c) << (2 * logW));
  const float* A  = a + plane;
  const float* Bp = bopt ? bopt + plane : nullptr;

  unsigned i00 = ((unsigned)y0 << logW) + x0, i01 = ((unsigned)y0 << logW) + x1;
  unsigned i10 = ((unsigned)y1 << logW) + x0, i11 = ((unsigned)y1 << logW) + x1;
  float v00 = A[i00] + (Bp ? Bp[i00] : 0.0f);
  float v01 = A[i01] + (Bp ? Bp[i01] : 0.0f);
  float v10 = A[i10] + (Bp ? Bp[i10] : 0.0f);
  float v11 = A[i11] + (Bp ? Bp[i11] : 0.0f);

  float v = v00 * (1.0f - fy) * (1.0f - fx) + v01 * (1.0f - fy) * fx +
            v10 * fy * (1.0f - fx) + v11 * fy * fx;
  if (relu) v = fmaxf(v, 0.0f);
  out[idx] = v;
}

// ---------------------------------------------------------------------------
// Host orchestration
// ---------------------------------------------------------------------------
extern "C" void kernel_launch(void* const* d_in, const int* in_sizes, int n_in,
                              void* d_out, int out_size, void* d_ws, size_t ws_size,
                              hipStream_t stream) {
  (void)in_sizes; (void)n_in; (void)out_size; (void)ws_size;
  const int B = 8;

  // --- inputs (JAX pytree / sorted-key flattening) ---
  const float* feat_test0  = (const float*)d_in[0];  // (8,64,256,256)
  const float* feat_test1  = (const float*)d_in[1];  // (8,256,128,128)
  const float* feat_test2  = (const float*)d_in[2];  // (8,512,64,64)
  const float* feat_test3  = (const float*)d_in[3];  // (8,1024,32,32)
  const float* feat_train3 = (const float*)d_in[4];  // (8,1024,32,32)
  const float* mask_train  = (const float*)d_in[5];  // (8,1,512,512)

  // params: sorted layers, each {b, be, g, rm, rv, w}
  // 0 f0_0  1 f0_1  2 f1_0  3 f1_1  4 f2_0  5 f2_1  6 mixer0  7 mixer1
  // 8 post0_0  9 post0_1 10 post1_0 11 post1_1 12 post2_0 13 post2_1
  // 14 s3_0 15 s3_1 16 segment0 17 segment1
  auto conv = [&](const float* in, int layer, float* out, int ci, int co,
                  int logW, int ksz, int pad, int relu) {
    int base = 6 + layer * 6;
    const float* b_  = (const float*)d_in[base + 0];
    const float* be_ = (const float*)d_in[base + 1];
    const float* g_  = (const float*)d_in[base + 2];
    const float* rm_ = (const float*)d_in[base + 3];
    const float* rv_ = (const float*)d_in[base + 4];
    const float* w_  = (const float*)d_in[base + 5];
    int HW = 1 << (2 * logW);
    dim3 grid(HW / 64, (co + 15) / 16, B);
    if (ksz == 1)
      convbn_wmma<1><<<grid, 128, 0, stream>>>(in, w_, b_, g_, be_, rm_, rv_, out,
                                               ci, co, logW, pad, relu);
    else
      convbn_wmma<3><<<grid, 128, 0, stream>>>(in, w_, b_, g_, be_, rm_, rv_, out,
                                               ci, co, logW, pad, relu);
  };
  auto up2 = [&](const float* a, const float* badd, float* out, int logC, int logW,
                 int relu) {
    size_t total = (size_t)B << (logC + 2 * (logW + 1));
    up2_kernel<<<(total + 255) / 256, 256, 0, stream>>>(a, badd, out, logC, logW, relu);
  };

  // --- workspace layout (floats); peak ~25.2M floats (~101 MB) ---
  float* F = (float*)d_ws;
  float* X1 = F + 0;                 //  524288
  float* X2 = F + 524288;            //  524288
  float* X3 = F + 1048576;           //  524288
  unsigned int* H1 = (unsigned int*)(F + 1572864);  // 262144 u32 (ftest packed)
  unsigned int* H2 = (unsigned int*)(F + 1835008);  // 262144 u32 (ftrain packed)
  float* S  = F + 2097152;           //   16384  segm_layers
  float* U  = F + 2113536;           // 2097152
  float* G  = F + 4210688;           // 1048576
  float* O3 = F + 5259264;           // 1048576
  float* U2 = F + 6307840;           // 4194304
  float* P  = F + 10502144;          // 2097152
  float* O2 = F + 12599296;          // 2097152
  float* U1 = F + 14696448;          // 8388608
  float* O1 = F + 23085056;          // 2097152

  // --- stems: segment0 (1x1, 1024->64) + segment1 (3x3, 64->64), 32x32 (log 5) ---
  conv(feat_test3, 16, X1, 1024, 64, 5, 1, 0, 1);   // t3a
  conv(X1,         17, X2,   64, 64, 5, 3, 1, 0);   // f_test
  conv(feat_train3,16, X1, 1024, 64, 5, 1, 0, 1);   // r3a
  conv(X1,         17, X3,   64, 64, 5, 3, 1, 0);   // f_train

  // --- l2-normalize + f16 pack ---
  l2pack<<<32, 256, 0, stream>>>(X2, H1, 1);  // ftest -> A layout [b][q][c/2]
  l2pack<<<32, 256, 0, stream>>>(X3, H2, 0);  // ftrain -> B layout [b][c/2][t]

  // --- similarity GEMM + masked top-3 + softmax -> segm_layers (B,2,32,32) ---
  {
    dim3 grid(16, B);
    simtopk_kernel<<<grid, 128, 0, stream>>>(H1, H2, mask_train, S);
  }

  // --- mixer + s3 ---
  conv(S,  6, X1,  2, 32, 5, 3, 1, 1);              // m0
  conv(X1, 7, X2, 32, 64, 5, 3, 1, 0);              // m1
  up2(X2, nullptr, U, 6, 5, 0);                     // u3 (8,64,64,64)
  conv(U, 14, G,  64, 32, 6, 3, 1, 1);              // s30
  conv(G, 15, O3, 32, 32, 6, 3, 1, 0);              // out3

  // --- f2 branch + post2 ---
  conv(feat_test2, 4, U,  512, 64, 6, 3, 1, 1);     // f2a
  conv(U,          5, G,   64, 32, 6, 3, 1, 0);     // f2b
  up2(G, O3, U2, 5, 6, 1);                          // u2 = relu(up2(f2b+out3))
  conv(U2, 12, P,  32, 16, 7, 3, 1, 1);             // p20
  conv(P,  13, O2, 16, 16, 7, 3, 1, 0);             // out2

  // --- f1 branch + post1 ---
  conv(feat_test1, 2, U2, 256, 32, 7, 3, 1, 1);     // f1a
  conv(U2,         3, P,   32, 16, 7, 3, 1, 0);     // f1b
  up2(P, O2, U1, 4, 7, 1);                          // u1 (8,16,256,256)
  conv(U1, 10, P,  16, 4, 8, 3, 1, 1);              // p10
  conv(P,  11, O1,  4, 4, 8, 3, 1, 0);              // out1

  // --- f0 branch + post0 ---
  conv(feat_test0, 0, U1, 64, 16, 8, 3, 1, 1);      // f0a
  conv(U1,         1, P,  16,  4, 8, 3, 1, 0);      // f0b
  up2(P, O1, U1, 2, 8, 1);                          // u0 (8,4,512,512)
  conv(U1, 8, U2, 4, 2, 9, 3, 1, 1);                // p00
  conv(U2, 9, (float*)d_out, 2, 2, 9, 3, 1, 0);     // final (8,2,512,512)
}
